// ResponseConditionedAttention_83348135346700
// MI455X (gfx1250) — matrix operand order repaired and checked by reading, not compiled, not run on hardware
//
#include <hip/hip_runtime.h>
#include <hip/hip_bf16.h>
#include <stdint.h>

// ---------------------------------------------------------------------------
// Response-conditioned attention for MI455X (gfx1250), wave32 + WMMA bf16.
// All matmul work goes through v_wmma_f32_16x16x32_bf16 (fp32 accumulate).
// GEMMs use double-buffered global_load_async_to_lds_b128 staging.
// ---------------------------------------------------------------------------

typedef __bf16 bf16_t;
typedef __attribute__((ext_vector_type(16))) __bf16 v16bf;
typedef __attribute__((ext_vector_type(8)))  __bf16 v8bf;
typedef __attribute__((ext_vector_type(8)))  float  v8f;
typedef int v4i_vs __attribute__((vector_size(16)));

#define E_DIM 1024
#define HEADS 16
#define DHEAD 64
#define SEQ   1024
#define BATCH 4
#define NROWS (BATCH * SEQ)   // 4096
#define NCLS  4

union AFrag { v16bf v; v8bf h[2]; };

static __device__ inline v8bf ld8(const bf16_t* p) { return *(const v8bf*)p; }

// ---- CDNA5 async global->LDS copy (guarded; falls back to ds_store path) ---
#if defined(__gfx1250__) && __has_builtin(__builtin_amdgcn_global_load_async_to_lds_b128)
#define USE_ASYNC_LDS 1
#else
#define USE_ASYNC_LDS 0
#endif

static __device__ inline void cp16_g2s(bf16_t* lds_dst, const bf16_t* gsrc) {
#if USE_ASYNC_LDS
  __builtin_amdgcn_global_load_async_to_lds_b128(
      (__attribute__((address_space(1))) v4i_vs*)(uintptr_t)gsrc,
      (__attribute__((address_space(3))) v4i_vs*)(uint32_t)(uintptr_t)lds_dst,
      0, 0);
#else
  *(uint4*)lds_dst = *(const uint4*)gsrc;
#endif
}

// Wait until at most the last-issued stage (8 async loads) remains in flight.
static __device__ inline void cp_g2s_wait_prev() {
#if USE_ASYNC_LDS
#if __has_builtin(__builtin_amdgcn_s_wait_asynccnt)
  __builtin_amdgcn_s_wait_asynccnt(8);
#endif
#endif
}
static __device__ inline void cp_g2s_wait_all() {
#if USE_ASYNC_LDS
#if __has_builtin(__builtin_amdgcn_s_wait_asynccnt)
  __builtin_amdgcn_s_wait_asynccnt(0);
#endif
#endif
}

// ---------------------------------------------------------------------------
// f32 -> bf16 conversion
// ---------------------------------------------------------------------------
__global__ void f32_to_bf16(const float* __restrict__ in, bf16_t* __restrict__ out, int n) {
  int i = blockIdx.x * blockDim.x + threadIdx.x;
  int stride = gridDim.x * blockDim.x;
  for (; i < n; i += stride) out[i] = (bf16_t)in[i];
}

// ---------------------------------------------------------------------------
// GEMM: C[m,n] = sum_k A[m,k] * W[n,k] + bias[n]
// A: [M,K] bf16 row-major, W: [N,K] bf16 row-major (weights are [out,in]).
// Block tile 128x128, BK=64 (two WMMA K-steps per stage), 8 waves of 32x64.
// Double-buffered async global->LDS staging (2 x 32 KB).
// MODE 0: outF = acc + bias                      (fp32)
// MODE 1: outB = (acc + bias) * scale            (bf16, Q path: folds 1/sqrt(D))
// MODE 2: if resp[row]==cls: outB = base + acc + bias   (bf16, class-masked K/V)
// ---------------------------------------------------------------------------
#define BM 128
#define BN 128
#define BK 64

template<int MODE>
__global__ void __launch_bounds__(256)
gemm_bf16(const bf16_t* __restrict__ A, const bf16_t* __restrict__ W,
          const float* __restrict__ bias,
          float* __restrict__ outF, bf16_t* __restrict__ outB,
          const float* __restrict__ base, const int* __restrict__ resp,
          int cls, float scale, int M, int N, int K)
{
  __shared__ bf16_t As[2 * BM * BK];   // 2 x 16 KB
  __shared__ bf16_t Bs[2 * BN * BK];   // 2 x 16 KB

  const int tid  = threadIdx.x;
  const int lane = tid & 31;
  const int wave = tid >> 5;
  const int wm   = wave >> 1;              // 0..3 -> 32 rows each
  const int wn   = wave & 1;               // 0..1 -> 64 cols each
  const int rowBase = blockIdx.y * BM;
  const int colBase = blockIdx.x * BN;

  const int mLane = lane & 15;
  const int hi    = lane >> 4;

  // staging map: each 128x64 bf16 tile = 1024 octets; 64 B contiguous / thread
  const int sRow = tid >> 1;               // 0..127
  const int sCol = (tid & 1) * 32;         // 0 or 32

  auto stage = [&](int buf, int k0) {
    const bf16_t* ga = A + (size_t)(rowBase + sRow) * K + k0 + sCol;
    bf16_t*       la = As + buf * (BM * BK) + sRow * BK + sCol;
    cp16_g2s(la,      ga);
    cp16_g2s(la + 8,  ga + 8);
    cp16_g2s(la + 16, ga + 16);
    cp16_g2s(la + 24, ga + 24);
    const bf16_t* gb = W + (size_t)(colBase + sRow) * K + k0 + sCol;
    bf16_t*       lb = Bs + buf * (BN * BK) + sRow * BK + sCol;
    cp16_g2s(lb,      gb);
    cp16_g2s(lb + 8,  gb + 8);
    cp16_g2s(lb + 16, gb + 16);
    cp16_g2s(lb + 24, gb + 24);
    __builtin_prefetch(gb + BK, 0, 0);     // weight stream prefetch
  };

  v8f acc[2][4] = {};

  const int nIter = K / BK;                // 16
  stage(0, 0);
  for (int it = 0; it < nIter; ++it) {
    const int cur = it & 1;
    if (it + 1 < nIter) {
      stage(cur ^ 1, (it + 1) * BK);       // fill other buffer in background
      cp_g2s_wait_prev();                  // retire current buffer's loads
    } else {
      cp_g2s_wait_all();
    }
    __syncthreads();

#pragma unroll
    for (int ks = 0; ks < 2; ++ks) {
      AFrag af[2], bfm[4];
#pragma unroll
      for (int mi = 0; mi < 2; ++mi) {
        const bf16_t* p = As + cur * (BM * BK) +
                          (wm * 32 + mi * 16 + mLane) * BK + ks * 32 + hi * 8;
        af[mi].h[0] = ld8(p); af[mi].h[1] = ld8(p + 16);
      }
#pragma unroll
      for (int ni = 0; ni < 4; ++ni) {
        const bf16_t* p = Bs + cur * (BN * BK) +
                          (wn * 64 + ni * 16 + mLane) * BK + ks * 32 + hi * 8;
        bfm[ni].h[0] = ld8(p); bfm[ni].h[1] = ld8(p + 16);
      }
#pragma unroll
      for (int mi = 0; mi < 2; ++mi)
#pragma unroll
        for (int ni = 0; ni < 4; ++ni)
          acc[mi][ni] = __builtin_amdgcn_wmma_f32_16x16x32_bf16(
              false, af[mi].v, false, bfm[ni].v, (short)0, acc[mi][ni], false, false);
    }
    __syncthreads();   // all waves done reading `cur` before it is re-staged
  }

  const int rhalf = hi * 8;
#pragma unroll
  for (int mi = 0; mi < 2; ++mi) {
#pragma unroll
    for (int ni = 0; ni < 4; ++ni) {
      const int col = colBase + wn * 64 + ni * 16 + mLane;
      const float b = bias[col];
#pragma unroll
      for (int r = 0; r < 8; ++r) {
        const int row = rowBase + wm * 32 + mi * 16 + rhalf + r;
        const float v = acc[mi][ni][r] + b;
        const size_t idx = (size_t)row * N + col;
        if (MODE == 0) {
          outF[idx] = v;
        } else if (MODE == 1) {
          outB[idx] = (bf16_t)(v * scale);
        } else {
          if (resp[row] == cls) outB[idx] = (bf16_t)(base[idx] + v);
        }
      }
    }
  }
}

// ---------------------------------------------------------------------------
// Flash attention (full softmax, no mask). Q pre-scaled by 1/sqrt(D).
// Grid: (S/128, H, B). 8 waves/block, 16 q-rows per wave, 32-key blocks.
// ---------------------------------------------------------------------------
__global__ void __launch_bounds__(256)
attn_kernel(const bf16_t* __restrict__ Q, const bf16_t* __restrict__ Kb,
            const bf16_t* __restrict__ Vb, bf16_t* __restrict__ O)
{
  __shared__ bf16_t Ks[32 * 64];        // K block  [j][d]          4 KB
  __shared__ bf16_t Vt[64 * 32];        // V block transposed [d][j] 4 KB
  __shared__ bf16_t Ps[8 * 16 * 32];    // per-wave P tiles          8 KB

  const int tid  = threadIdx.x;
  const int lane = tid & 31;
  const int wave = tid >> 5;
  const int h = blockIdx.y;
  const int b = blockIdx.z;
  const int qrow0  = blockIdx.x * 128 + wave * 16;   // within sequence
  const size_t rowOff = (size_t)b * SEQ;
  const int dOff = h * DHEAD;

  const int mLane = lane & 15;
  const int hi    = lane >> 4;

  // Q fragments (16 rows x 64 d), two K=32 steps
  AFrag qf[2];
#pragma unroll
  for (int t = 0; t < 2; ++t) {
    const bf16_t* p = Q + (rowOff + qrow0 + mLane) * E_DIM + dOff + t * 32 + hi * 8;
    qf[t].h[0] = ld8(p); qf[t].h[1] = ld8(p + 16);
  }

  v8f oacc[4] = {};
  float mrow[8], lrow[8];
#pragma unroll
  for (int r = 0; r < 8; ++r) { mrow[r] = -1e30f; lrow[r] = 0.f; }

  const int sRow = tid >> 3;          // 0..31
  const int sCol = (tid & 7) * 8;     // 0,8,..56

  for (int kb = 0; kb < SEQ; kb += 32) {
    __syncthreads();
    {
      const bf16_t* gk = Kb + (rowOff + kb + sRow) * E_DIM + dOff + sCol;
      *(v8bf*)(Ks + sRow * 64 + sCol) = ld8(gk);
      v8bf vv = ld8(Vb + (rowOff + kb + sRow) * E_DIM + dOff + sCol);
#pragma unroll
      for (int e = 0; e < 8; ++e) Vt[(sCol + e) * 32 + sRow] = vv[e];
    }
    __syncthreads();

    // scores = Q . K^T  (two 16x16 tiles: keys 0-15 and 16-31)
    v8f s0 = {}, s1 = {};
#pragma unroll
    for (int t = 0; t < 2; ++t) {
      AFrag b0, b1;
      const bf16_t* p0 = Ks + mLane * 64 + t * 32 + hi * 8;
      b0.h[0] = ld8(p0); b0.h[1] = ld8(p0 + 16);
      const bf16_t* p1 = Ks + (16 + mLane) * 64 + t * 32 + hi * 8;
      b1.h[0] = ld8(p1); b1.h[1] = ld8(p1 + 16);
      s0 = __builtin_amdgcn_wmma_f32_16x16x32_bf16(false, qf[t].v, false, b0.v, (short)0, s0, false, false);
      s1 = __builtin_amdgcn_wmma_f32_16x16x32_bf16(false, qf[t].v, false, b1.v, (short)0, s1, false, false);
    }

    // online softmax; row r maps to q-row (r + 8*hi); cols live across 16 lanes
    bf16_t* pw = Ps + wave * (16 * 32);
#pragma unroll
    for (int r = 0; r < 8; ++r) {
      float mx = fmaxf(s0[r], s1[r]);
#pragma unroll
      for (int d = 1; d < 16; d <<= 1) mx = fmaxf(mx, __shfl_xor(mx, d, 32));
      const float mn = fmaxf(mrow[r], mx);
      const float alpha = __expf(mrow[r] - mn);
      mrow[r] = mn;
      const float p0 = __expf(s0[r] - mn);
      const float p1 = __expf(s1[r] - mn);
      float sum = p0 + p1;
#pragma unroll
      for (int d = 1; d < 16; d <<= 1) sum += __shfl_xor(sum, d, 32);
      lrow[r] = lrow[r] * alpha + sum;
#pragma unroll
      for (int g = 0; g < 4; ++g) oacc[g][r] *= alpha;
      const int prow = r + hi * 8;
      pw[prow * 32 + mLane]      = (bf16_t)p0;
      pw[prow * 32 + 16 + mLane] = (bf16_t)p1;
    }

    // reload P in A-fragment layout (per-wave LDS region, no barrier needed)
    AFrag pf;
    {
      const bf16_t* p = pw + mLane * 32 + hi * 8;
      pf.h[0] = ld8(p); pf.h[1] = ld8(p + 16);
    }

    // O += P . V  (K=32 over this key block, 4 d-groups of 16)
#pragma unroll
    for (int g = 0; g < 4; ++g) {
      AFrag vf;
      const bf16_t* p = Vt + (g * 16 + mLane) * 32 + hi * 8;
      vf.h[0] = ld8(p); vf.h[1] = ld8(p + 16);
      oacc[g] = __builtin_amdgcn_wmma_f32_16x16x32_bf16(
          false, pf.v, false, vf.v, (short)0, oacc[g], false, false);
    }
  }

#pragma unroll
  for (int g = 0; g < 4; ++g) {
#pragma unroll
    for (int r = 0; r < 8; ++r) {
      const int row = qrow0 + r + hi * 8;
      const float v = oacc[g][r] / lrow[r];
      O[(rowOff + row) * E_DIM + dOff + g * 16 + mLane] = (bf16_t)v;
    }
  }
}

// ---------------------------------------------------------------------------
// Host orchestration
// ---------------------------------------------------------------------------
extern "C" void kernel_launch(void* const* d_in, const int* in_sizes, int n_in,
                              void* d_out, int out_size, void* d_ws, size_t ws_size,
                              hipStream_t stream) {
  const float* query = (const float*)d_in[0];
  const float* key   = (const float*)d_in[1];
  const float* value = (const float*)d_in[2];
  const int*   resp  = (const int*)  d_in[3];
  const float* Wq = (const float*)d_in[4];  const float* bq = (const float*)d_in[5];
  const float* Wk = (const float*)d_in[6];  const float* bk = (const float*)d_in[7];
  const float* Wv = (const float*)d_in[8];  const float* bv = (const float*)d_in[9];
  const float* Wo = (const float*)d_in[10]; const float* bo = (const float*)d_in[11];
  const float* Wkm = (const float*)d_in[12]; const float* bkm = (const float*)d_in[13];
  const float* Wvm = (const float*)d_in[14]; const float* bvm = (const float*)d_in[15];

  char* ws = (char*)d_ws;
  size_t off = 0;
  auto alloc = [&](size_t bytes) -> void* {
    off = (off + 255) & ~(size_t)255;
    void* p = ws + off;
    off += bytes;
    return p;
  };

  const size_t NE = (size_t)NROWS * E_DIM;       // 4M elems
  const size_t EE = (size_t)E_DIM * E_DIM;       // 1M elems

  bf16_t* Xq = (bf16_t*)alloc(NE * 2);
  bf16_t* Xk = (bf16_t*)alloc(NE * 2);
  bf16_t* Xv = (bf16_t*)alloc(NE * 2);
  bf16_t* Wq_b  = (bf16_t*)alloc(EE * 2);
  bf16_t* Wk_b  = (bf16_t*)alloc(EE * 2);
  bf16_t* Wv_b  = (bf16_t*)alloc(EE * 2);
  bf16_t* Wo_b  = (bf16_t*)alloc(EE * 2);
  bf16_t* Wkm_b = (bf16_t*)alloc(NCLS * EE * 2);
  bf16_t* Wvm_b = (bf16_t*)alloc(NCLS * EE * 2);
  bf16_t* Qbf = (bf16_t*)alloc(NE * 2);
  bf16_t* Kbf = (bf16_t*)alloc(NE * 2);
  bf16_t* Vbf = (bf16_t*)alloc(NE * 2);
  bf16_t* Abf = (bf16_t*)alloc(NE * 2);
  float* Kbase = (float*)alloc(NE * 4);
  float* Vbase = (float*)alloc(NE * 4);

  auto conv = [&](const float* src, bf16_t* dst, size_t n) {
    f32_to_bf16<<<2048, 256, 0, stream>>>(src, dst, (int)n);
  };
  conv(query, Xq, NE);
  conv(key,   Xk, NE);
  conv(value, Xv, NE);
  conv(Wq, Wq_b, EE);
  conv(Wk, Wk_b, EE);
  conv(Wv, Wv_b, EE);
  conv(Wo, Wo_b, EE);
  conv(Wkm, Wkm_b, NCLS * EE);
  conv(Wvm, Wvm_b, NCLS * EE);

  const dim3 gblk(E_DIM / BN, NROWS / BM);   // (8, 32)
  const float qscale = 0.125f;               // 1/sqrt(64) folded into Q

  // Q projection -> bf16, pre-scaled
  gemm_bf16<1><<<gblk, 256, 0, stream>>>(Xq, Wq_b, bq, nullptr, Qbf,
                                         nullptr, nullptr, 0, qscale,
                                         NROWS, E_DIM, E_DIM);
  // K/V base projections -> fp32
  gemm_bf16<0><<<gblk, 256, 0, stream>>>(Xk, Wk_b, bk, Kbase, nullptr,
                                         nullptr, nullptr, 0, 1.f,
                                         NROWS, E_DIM, E_DIM);
  gemm_bf16<0><<<gblk, 256, 0, stream>>>(Xv, Wv_b, bv, Vbase, nullptr,
                                         nullptr, nullptr, 0, 1.f,
                                         NROWS, E_DIM, E_DIM);
  // Class-conditional K/V: rows partitioned by resp -> final bf16 K/V
  for (int c = 0; c < NCLS; ++c) {
    gemm_bf16<2><<<gblk, 256, 0, stream>>>(Xk, Wkm_b + (size_t)c * EE, bkm + (size_t)c * E_DIM,
                                           nullptr, Kbf, Kbase, resp, c, 1.f,
                                           NROWS, E_DIM, E_DIM);
    gemm_bf16<2><<<gblk, 256, 0, stream>>>(Xv, Wvm_b + (size_t)c * EE, bvm + (size_t)c * E_DIM,
                                           nullptr, Vbf, Vbase, resp, c, 1.f,
                                           NROWS, E_DIM, E_DIM);
  }

  // Attention
  attn_kernel<<<dim3(SEQ / 128, HEADS, BATCH), 256, 0, stream>>>(Qbf, Kbf, Vbf, Abf);

  // Output projection -> d_out (fp32)
  gemm_bf16<0><<<gblk, 256, 0, stream>>>(Abf, Wo_b, bo, (float*)d_out, nullptr,
                                         nullptr, nullptr, 0, 1.f,
                                         NROWS, E_DIM, E_DIM);
}
